// KVCacheTransformer_88682484728138
// MI455X (gfx1250) — matrix operand order repaired and checked by reading, not compiled, not run on hardware
//
#include <hip/hip_runtime.h>

// MI455X / gfx1250, wave32. Memory-bandwidth-bound kernel (~4.3 GB HBM traffic
// => ~185us floor @ 23.3 TB/s). K/V streams are each touched exactly once:
// the concat-copy is fused with the attention score / attn*V passes.

typedef __attribute__((ext_vector_type(16))) __bf16 v16bf;
typedef __attribute__((ext_vector_type(8)))  float  v8f;
typedef __attribute__((ext_vector_type(4)))  float  v4f;

#define DM     1024
#define NB     64
#define KVLEN  4096
#define STOT   4097   // KVLEN + 1 new token

// Pack 4x v4f (16 floats) into v16bf, element i <- chunk order given.
__device__ __forceinline__ v16bf pack16(v4f a0, v4f a1, v4f a2, v4f a3) {
  v16bf r;
  r[0]=(__bf16)a0.x;  r[1]=(__bf16)a0.y;  r[2]=(__bf16)a0.z;  r[3]=(__bf16)a0.w;
  r[4]=(__bf16)a1.x;  r[5]=(__bf16)a1.y;  r[6]=(__bf16)a1.z;  r[7]=(__bf16)a1.w;
  r[8]=(__bf16)a2.x;  r[9]=(__bf16)a2.y;  r[10]=(__bf16)a2.z; r[11]=(__bf16)a2.w;
  r[12]=(__bf16)a3.x; r[13]=(__bf16)a3.y; r[14]=(__bf16)a3.z; r[15]=(__bf16)a3.w;
  return r;
}

// ---------------------------------------------------------------------------
// 1) Embedding gather: cur[b][:] = emb[x[b]][:]
__global__ __launch_bounds__(256) void gather_embed(
    const int* __restrict__ x, const float* __restrict__ emb,
    float* __restrict__ cur) {
  int b = blockIdx.x;
  int d = threadIdx.x * 4;
  size_t row = (size_t)x[b];
  *(v4f*)(cur + (size_t)b * DM + d) = *(const v4f*)(emb + row * DM + d);
}

// ---------------------------------------------------------------------------
// 2) Projection GEMM  Out[m][e] = sum_d A[m][d] * W[e][d] + bias[e]
// M=64, N=K=1024. One wave per 16x16 tile, WMMA f32_16x16x32_bf16.
// A layout (16-bit A 16x32): lane<16: M=lane, K in {0-7,16-23};
//                            lane>=16: M=lane-16, K in {8-15,24-31}.
// B layout (16-bit B 32x16): N=lane&15, K as for A (lane<16: {0-7,16-23}; lane>=16: {8-15,24-31}).
__global__ __launch_bounds__(32) void gemm_proj_wmma(
    const float* __restrict__ A, const float* __restrict__ W,
    const float* __restrict__ bias, float* __restrict__ Out) {
  int n0 = blockIdx.x * 16;
  int m0 = blockIdx.y * 16;
  int lane = threadIdx.x;
  int l15  = lane & 15;
  int offA = (lane < 16) ? 0 : 8;
  int offB = (lane < 16) ? 0 : 8;
  const float* arow = A + (size_t)(m0 + l15) * DM;
  const float* wrow = W + (size_t)(n0 + l15) * DM;   // W[e][d]: row e is B column
  v8f acc = {};
  for (int kk = 0; kk < DM; kk += 32) {
    int da = kk + offA;
    v4f a0 = *(const v4f*)(arow + da);
    v4f a1 = *(const v4f*)(arow + da + 4);
    v4f a2 = *(const v4f*)(arow + da + 16);
    v4f a3 = *(const v4f*)(arow + da + 20);
    int db = kk + offB;
    v4f b0 = *(const v4f*)(wrow + db);
    v4f b1 = *(const v4f*)(wrow + db + 4);
    v4f b2 = *(const v4f*)(wrow + db + 16);
    v4f b3 = *(const v4f*)(wrow + db + 20);
    v16bf av = pack16(a0, a1, a2, a3);
    v16bf bv = pack16(b0, b1, b2, b3);
    acc = __builtin_amdgcn_wmma_f32_16x16x32_bf16(
        false, av, false, bv, (short)0, acc, false, false);
  }
  // D layout: VGPR r, lane<16 -> M=r ; lane>=16 -> M=r+8 ; N = lane&15
  int col   = n0 + l15;
  int rbase = m0 + ((lane < 16) ? 0 : 8);
  float bc = bias[col];
  for (int r = 0; r < 8; ++r)
    Out[(size_t)(rbase + r) * DM + col] = acc[r] + bc;
}

// ---------------------------------------------------------------------------
// 3) Copy k_new/v_new into output row s=4096
__global__ __launch_bounds__(256) void copy_new_kv(
    const float* __restrict__ k_new, const float* __restrict__ v_new,
    float* __restrict__ out_k, float* __restrict__ out_v) {
  int b = blockIdx.x;
  int d = threadIdx.x * 4;
  size_t o = ((size_t)b * STOT + KVLEN) * DM + d;
  *(v4f*)(out_k + o) = *(const v4f*)(k_new + (size_t)b * DM + d);
  *(v4f*)(out_v + o) = *(const v4f*)(v_new + (size_t)b * DM + d);
}

// ---------------------------------------------------------------------------
// 4) Fused: stream prev_k -> out_k (NT), and scores[b][s] = q.k / 32 via WMMA.
// A tile = 16 rows of K (the lanes of the A layout collectively cover every
// element of the 16x32 block exactly once -> the copy is free).
// B tile = q chunk replicated in every column; read result from column 0.
__global__ __launch_bounds__(32) void score_copyk_wmma(
    const float* __restrict__ prev_k, const float* __restrict__ q,
    float* __restrict__ out_k, float* __restrict__ scores) {
  int b  = blockIdx.x;
  int s0 = blockIdx.y * 16;
  int lane = threadIdx.x;
  int l15  = lane & 15;
  int offA = (lane < 16) ? 0 : 8;
  int offB = (lane < 16) ? 0 : 8;
  const float* krow = prev_k + ((size_t)b * KVLEN + s0 + l15) * DM;
  float*       orow = out_k  + ((size_t)b * STOT  + s0 + l15) * DM;
  const float* qb   = q + (size_t)b * DM;
  v8f acc = {};
  for (int kk = 0; kk < DM; kk += 32) {
    int da = kk + offA;
    v4f a0 = __builtin_nontemporal_load((const v4f*)(krow + da));
    v4f a1 = __builtin_nontemporal_load((const v4f*)(krow + da + 4));
    v4f a2 = __builtin_nontemporal_load((const v4f*)(krow + da + 16));
    v4f a3 = __builtin_nontemporal_load((const v4f*)(krow + da + 20));
    __builtin_nontemporal_store(a0, (v4f*)(orow + da));
    __builtin_nontemporal_store(a1, (v4f*)(orow + da + 4));
    __builtin_nontemporal_store(a2, (v4f*)(orow + da + 16));
    __builtin_nontemporal_store(a3, (v4f*)(orow + da + 20));
    int db = kk + offB;
    v4f b0 = *(const v4f*)(qb + db);
    v4f b1 = *(const v4f*)(qb + db + 4);
    v4f b2 = *(const v4f*)(qb + db + 16);
    v4f b3 = *(const v4f*)(qb + db + 20);
    v16bf av = pack16(a0, a1, a2, a3);
    v16bf bv = pack16(b0, b1, b2, b3);
    acc = __builtin_amdgcn_wmma_f32_16x16x32_bf16(
        false, av, false, bv, (short)0, acc, false, false);
  }
  // every column of D is identical; lane 0 holds rows 0..7, lane 16 rows 8..15
  if (l15 == 0) {
    int mofs = (lane < 16) ? 0 : 8;
    const float inv = 1.0f / 32.0f;                 // 1/sqrt(1024)
    for (int r = 0; r < 8; ++r)
      scores[(size_t)b * STOT + s0 + mofs + r] = acc[r] * inv;
  }
}

// ---------------------------------------------------------------------------
// 5) Per-batch: last score (q . k_new) + softmax in place over 4097 scores.
__global__ __launch_bounds__(256) void softmax_kernel(
    const float* __restrict__ q, const float* __restrict__ k_new,
    float* __restrict__ scores) {
  __shared__ float red[256];
  int b = blockIdx.x, t = threadIdx.x;
  const float* qb = q + (size_t)b * DM;
  const float* kb = k_new + (size_t)b * DM;
  float* sb = scores + (size_t)b * STOT;

  float p = 0.f;
  for (int d = t; d < DM; d += 256) p += qb[d] * kb[d];
  red[t] = p; __syncthreads();
  for (int st = 128; st > 0; st >>= 1) {
    if (t < st) red[t] += red[t + st];
    __syncthreads();
  }
  if (t == 0) sb[KVLEN] = red[0] * (1.0f / 32.0f);
  __syncthreads();

  float m = -3.0e38f;
  for (int i = t; i < STOT; i += 256) m = fmaxf(m, sb[i]);
  red[t] = m; __syncthreads();
  for (int st = 128; st > 0; st >>= 1) {
    if (t < st) red[t] = fmaxf(red[t], red[t + st]);
    __syncthreads();
  }
  m = red[0]; __syncthreads();

  float sum = 0.f;
  for (int i = t; i < STOT; i += 256) {
    float e = expf(sb[i] - m);
    sb[i] = e;
    sum += e;
  }
  red[t] = sum; __syncthreads();
  for (int st = 128; st > 0; st >>= 1) {
    if (t < st) red[t] += red[t + st];
    __syncthreads();
  }
  float inv = 1.0f / red[0];
  for (int i = t; i < STOT; i += 256) sb[i] *= inv;
}

// ---------------------------------------------------------------------------
// 6) Fused: stream prev_v -> out_v (NT) + partial attn = sum_s probs[s]*v[s].
// grid (b, dchunk 0..3, sseg 0..7), block 64: each block = 256 d x 512 s.
__global__ __launch_bounds__(64) void attnv_copyv(
    const float* __restrict__ prev_v, const float* __restrict__ probs,
    float* __restrict__ out_v, float* __restrict__ part) {
  int b = blockIdx.x, dc = blockIdx.y, seg = blockIdx.z;
  int d0 = dc * 256 + threadIdx.x * 4;
  const float* pb = probs + (size_t)b * STOT + seg * 512;
  const float* vb = prev_v + ((size_t)b * KVLEN + (size_t)seg * 512) * DM + d0;
  float*       ob = out_v  + ((size_t)b * STOT  + (size_t)seg * 512) * DM + d0;
  v4f acc = {};
  for (int s = 0; s < 512; ++s) {
    float p = pb[s];
    v4f v = __builtin_nontemporal_load((const v4f*)(vb + (size_t)s * DM));
    __builtin_nontemporal_store(v, (v4f*)(ob + (size_t)s * DM));
    acc += p * v;
  }
  v4f* po = (v4f*)(part + ((size_t)b * 8 + seg) * DM + d0);
  *po = acc;
}

// ---------------------------------------------------------------------------
// 7) attn[b][d] = sum_seg part + probs[4096]*v_new[b][d]
__global__ __launch_bounds__(256) void attn_reduce(
    const float* __restrict__ part, const float* __restrict__ probs,
    const float* __restrict__ v_new, float* __restrict__ attn) {
  int b = blockIdx.x;
  int d = threadIdx.x * 4;
  v4f acc = {};
  for (int seg = 0; seg < 8; ++seg)
    acc += *(const v4f*)(part + ((size_t)b * 8 + seg) * DM + d);
  float pl = probs[(size_t)b * STOT + KVLEN];
  acc += pl * *(const v4f*)(v_new + (size_t)b * DM + d);
  *(v4f*)(attn + (size_t)b * DM + d) = acc;
}

// ---------------------------------------------------------------------------
extern "C" void kernel_launch(void* const* d_in, const int* in_sizes, int n_in,
                              void* d_out, int out_size, void* d_ws, size_t ws_size,
                              hipStream_t stream) {
  const int*   x      = (const int*)  d_in[0];
  const float* prev_k = (const float*)d_in[1];
  const float* prev_v = (const float*)d_in[2];
  const float* emb    = (const float*)d_in[3];
  const float* Wq     = (const float*)d_in[4];
  const float* bq     = (const float*)d_in[5];
  const float* Wk     = (const float*)d_in[6];
  const float* bk     = (const float*)d_in[7];
  const float* Wv     = (const float*)d_in[8];
  const float* bv     = (const float*)d_in[9];
  const float* Wo     = (const float*)d_in[10];
  const float* bo     = (const float*)d_in[11];

  float* out   = (float*)d_out;                       // B x 1 x D
  float* out_k = out + (size_t)NB * DM;               // B x 4097 x D
  float* out_v = out_k + (size_t)NB * STOT * DM;      // B x 4097 x D

  // workspace layout (fp32), all offsets 16B-aligned
  char* ws = (char*)d_ws;
  float* cur    = (float*)(ws + 0);                   // 256 KB
  float* q      = (float*)(ws + (1u << 18));          // 256 KB
  float* k_new  = (float*)(ws + (2u << 18));          // 256 KB
  float* v_new  = (float*)(ws + (3u << 18));          // 256 KB
  float* scores = (float*)(ws + (4u << 18));          // 64*4097*4 ~ 1.05 MB
  float* part   = (float*)(ws + (4u << 18) + (2u << 20)); // 64*8*1024*4 = 2 MB
  float* attn   = (float*)(ws + (4u << 18) + (4u << 20)); // 256 KB
  (void)ws_size; (void)in_sizes; (void)n_in; (void)out_size;

  gather_embed<<<NB, 256, 0, stream>>>(x, emb, cur);

  dim3 gG(DM / 16, NB / 16);
  gemm_proj_wmma<<<gG, 32, 0, stream>>>(cur, Wq, bq, q);
  gemm_proj_wmma<<<gG, 32, 0, stream>>>(cur, Wk, bk, k_new);
  gemm_proj_wmma<<<gG, 32, 0, stream>>>(cur, Wv, bv, v_new);

  copy_new_kv<<<NB, 256, 0, stream>>>(k_new, v_new, out_k, out_v);

  score_copyk_wmma<<<dim3(NB, KVLEN / 16), 32, 0, stream>>>(prev_k, q, out_k, scores);

  softmax_kernel<<<NB, 256, 0, stream>>>(q, k_new, scores);

  attnv_copyv<<<dim3(NB, 4, 8), 64, 0, stream>>>(prev_v, scores, out_v, part);

  attn_reduce<<<NB, 256, 0, stream>>>(part, scores, v_new, attn);

  gemm_proj_wmma<<<gG, 32, 0, stream>>>(attn, Wo, bo, out);
}
